// DeltaAttention_78245714199328
// MI455X (gfx1250) — matrix operand order repaired and checked by reading, not compiled, not run on hardware
//
#include <hip/hip_runtime.h>

#define DEVI __device__ __forceinline__

typedef __attribute__((ext_vector_type(16))) _Float16 v16h;
typedef __attribute__((ext_vector_type(8)))  _Float16 v8h;
typedef __attribute__((ext_vector_type(8)))  float    v8f;

namespace {
constexpr int kB = 2;
constexpr int kT = 2048;
constexpr int kC = 1024;
constexpr int kH = 16;
constexpr int kD = 64;
constexpr int kN = kB * kT;   // 4096 rows
}

// A-operand fragment (16x32 f16, MxK): lane holds row m=lane&15,
// K elements {hi*8+0..7} in v0-3 and {hi*8+16..23} in v4-7 (hi = lane>>4).
DEVI v16h load_afrag(const _Float16* p, int lane) {
  const int hi = (lane >> 4) & 1;
  const _Float16* q = p + hi * 8;
  v8h a = *(const v8h*)(q);
  v8h b = *(const v8h*)(q + 16);
  v16h f;
#pragma unroll
  for (int i = 0; i < 8; ++i) { f[i] = a[i]; f[i + 8] = b[i]; }
  return f;
}

// B-operand fragment (32x16 f16, KxN): lane holds column n=lane&15,
// K elements {hi*8+0..7} in v0-3 and {hi*8+16..23} in v4-7 (hi = lane>>4).
DEVI v16h load_bfrag(const _Float16* p, int lane) {
  const int hi = (lane >> 4) & 1;
  const _Float16* q = p + hi * 8;
  v8h a = *(const v8h*)(q);
  v8h b = *(const v8h*)(q + 16);
  v16h f;
#pragma unroll
  for (int i = 0; i < 8; ++i) { f[i] = a[i]; f[i + 8] = b[i]; }
  return f;
}

DEVI v8f wmma_f16(v16h a, v16h b, v8f c) {
  return __builtin_amdgcn_wmma_f32_16x16x32_f16(false, a, false, b, (short)0, c,
                                                false, false);
}

// tanh (optional) + f32->f16 conversion + per-row sum of squares.
__global__ void k_prep(const float* __restrict__ src, _Float16* __restrict__ dsth,
                       float* __restrict__ rowsq, int cols, int do_tanh) {
  const int row = blockIdx.x;
  const int tid = threadIdx.x;
  const float* s = src + (size_t)row * cols;
  _Float16* d = dsth + (size_t)row * cols;
  float acc = 0.f;
  for (int i = tid; i < cols; i += blockDim.x) {
    float v = s[i];
    if (do_tanh) v = tanhf(v);
    d[i] = (_Float16)v;
    acc += v * v;
  }
  __shared__ float red[256];
  red[tid] = acc;
  __syncthreads();
  for (int off = 128; off > 0; off >>= 1) {
    if (tid < off) red[tid] += red[tid + off];
    __syncthreads();
  }
  if (tid == 0) rowsq[row] = red[0];
}

// Proxy-delta linear: dot = xa*wa^T via WMMA, then distance epilogue.
// mode: 0=q (tanh, [B,H,T,D]), 1=k (tanh, [B,H,T,D]), 2=v (no tanh, [B,H,D,T]).
__global__ void __launch_bounds__(32) k_qkv(
    const _Float16* __restrict__ xah, const float* __restrict__ xsq,
    const _Float16* __restrict__ wh, const float* __restrict__ wsq,
    const float* __restrict__ gain, int mode, _Float16* __restrict__ out) {
  const int lane = threadIdx.x;
  const int n = lane & 15;
  const int mbase = (lane >> 4) * 8;
  const int tm = blockIdx.x;   // row tile over N
  const int tn = blockIdx.y;   // col tile over C
  const _Float16* arow = xah + (size_t)(tm * 16 + n) * kC;
  const _Float16* brow = wh + (size_t)(tn * 16 + n) * kC;
  v8f acc = {};
  for (int k = 0; k < kC; k += 32) {
    acc = wmma_f16(load_afrag(arow + k, lane), load_bfrag(brow + k, lane), acc);
  }
  const int col = tn * 16 + n;
  const float wq = wsq[col];
  const float g = gain[col];
  const int h = col >> 6;      // /64
  const int d = col & 63;
#pragma unroll
  for (int r = 0; r < 8; ++r) {
    const int row = tm * 16 + mbase + r;
    float l2 = xsq[row] + wq - 2.f * acc[r];
    l2 = fmaxf(l2, 0.f);
    float val = (0.5f - l2 * (1.f / (float)kC)) * g;   // (1 - l2/Din - 0.5)*gain
    if (mode != 2) val = tanhf(val);
    const int bb = row >> 11;        // /T
    const int t = row & (kT - 1);
    if (mode == 2) {
      out[(((size_t)bb * kH + h) * kD + d) * kT + t] = (_Float16)val;  // V^T
    } else {
      out[(((size_t)bb * kH + h) * kT + t) * kD + d] = (_Float16)val;
    }
  }
}

// Flash-style distance attention. One wave per (b, h, 16-row q tile).
__global__ void __launch_bounds__(32) k_attn(const _Float16* __restrict__ qh,
                                             const _Float16* __restrict__ kh,
                                             const _Float16* __restrict__ vT,
                                             _Float16* __restrict__ aout) {
  const int lane = threadIdx.x;
  const int qt = blockIdx.x;
  const int h = blockIdx.y;
  const int b = blockIdx.z;
  const int n = lane & 15;
  const int mbase = (lane >> 4) * 8;
  const size_t bh = (size_t)b * kH + h;

  const _Float16* qrow = qh + (bh * kT + (size_t)qt * 16 + n) * kD;
  v16h aq0 = load_afrag(qrow, lane);        // K = 0..31
  v16h aq1 = load_afrag(qrow + 32, lane);   // K = 32..63

  // q row norms: lanes L and L^16 hold complementary halves of row (L&15).
  float myq = 0.f;
#pragma unroll
  for (int i = 0; i < 16; ++i) {
    float a = (float)aq0[i], c = (float)aq1[i];
    myq += a * a + c * c;
  }
  myq += __shfl_xor(myq, 16, 32);
  float qs[8];
#pragma unroll
  for (int r = 0; r < 8; ++r) qs[r] = __shfl(myq, mbase + r, 32);

  float M[8], Ls[8];
#pragma unroll
  for (int r = 0; r < 8; ++r) { M[r] = -3.0e38f; Ls[r] = 0.f; }
  v8f o0 = {}, o1 = {}, o2 = {}, o3 = {};

  __shared__ _Float16 pbuf[16 * 32];   // P tile staging (C-layout -> A-layout)

  const float inv80 = 1.f / 80.f;      // 1/(sqrt(64)*10)
  for (int st = 0; st < kT; st += 32) {
    const _Float16* krow0 = kh + (bh * kT + st + n) * kD;
    const _Float16* krow1 = krow0 + 16 * kD;
    v16h bk00 = load_bfrag(krow0, lane);
    v16h bk01 = load_bfrag(krow0 + 32, lane);
    v16h bk10 = load_bfrag(krow1, lane);
    v16h bk11 = load_bfrag(krow1 + 32, lane);

    v8f s0 = {}, s1 = {};
    s0 = wmma_f16(aq0, bk00, s0);
    s0 = wmma_f16(aq1, bk01, s0);
    s1 = wmma_f16(aq0, bk10, s1);
    s1 = wmma_f16(aq1, bk11, s1);

    float k0s = 0.f, k1s = 0.f;
#pragma unroll
    for (int i = 0; i < 16; ++i) {
      float a = (float)bk00[i], c = (float)bk01[i];
      k0s += a * a + c * c;
      float e = (float)bk10[i], g = (float)bk11[i];
      k1s += e * e + g * g;
    }
    k0s += __shfl_xor(k0s, 16, 32);
    k1s += __shfl_xor(k1s, 16, 32);

    float lg0[8], lg1[8], tmax[8];
#pragma unroll
    for (int r = 0; r < 8; ++r) {
      float d0 = fmaxf(qs[r] + k0s - 2.f * s0[r], 0.f);
      float d1 = fmaxf(qs[r] + k1s - 2.f * s1[r], 0.f);
      lg0[r] = -d0 * inv80;
      lg1[r] = -d1 * inv80;
      tmax[r] = fmaxf(lg0[r], lg1[r]);
    }
#pragma unroll
    for (int r = 0; r < 8; ++r) {   // row max across the 16 lanes of each half
      float t = tmax[r];
      t = fmaxf(t, __shfl_xor(t, 1, 32));
      t = fmaxf(t, __shfl_xor(t, 2, 32));
      t = fmaxf(t, __shfl_xor(t, 4, 32));
      t = fmaxf(t, __shfl_xor(t, 8, 32));
      tmax[r] = t;
    }
    float sc[8], p0[8], p1[8];
#pragma unroll
    for (int r = 0; r < 8; ++r) {
      float Mn = fmaxf(M[r], tmax[r]);
      sc[r] = __expf(M[r] - Mn);
      p0[r] = __expf(lg0[r] - Mn);
      p1[r] = __expf(lg1[r] - Mn);
      M[r] = Mn;
    }
#pragma unroll
    for (int r = 0; r < 8; ++r) {
      float t = p0[r] + p1[r];
      t += __shfl_xor(t, 1, 32);
      t += __shfl_xor(t, 2, 32);
      t += __shfl_xor(t, 4, 32);
      t += __shfl_xor(t, 8, 32);
      Ls[r] = Ls[r] * sc[r] + t;
    }
    // stage P (16x32) to LDS, reload in A layout
#pragma unroll
    for (int r = 0; r < 8; ++r) {
      pbuf[(mbase + r) * 32 + n] = (_Float16)p0[r];
      pbuf[(mbase + r) * 32 + n + 16] = (_Float16)p1[r];
    }
    __syncthreads();
    v16h ap = load_afrag(&pbuf[n * 32], lane);
    __syncthreads();
#pragma unroll
    for (int r = 0; r < 8; ++r) {
      o0[r] *= sc[r]; o1[r] *= sc[r]; o2[r] *= sc[r]; o3[r] *= sc[r];
    }
    const _Float16* vbase = vT + (bh * kD + n) * kT + st;   // V^T: contiguous in s
    o0 = wmma_f16(ap, load_bfrag(vbase, lane), o0);
    o1 = wmma_f16(ap, load_bfrag(vbase + (size_t)16 * kT, lane), o1);
    o2 = wmma_f16(ap, load_bfrag(vbase + (size_t)32 * kT, lane), o2);
    o3 = wmma_f16(ap, load_bfrag(vbase + (size_t)48 * kT, lane), o3);
  }

  const size_t nrow0 = (size_t)b * kT + (size_t)qt * 16 + mbase;
#pragma unroll
  for (int r = 0; r < 8; ++r) {
    float il = 1.f / Ls[r];
    size_t base = (nrow0 + r) * kC + (size_t)h * kD + n;
    aout[base]      = (_Float16)(o0[r] * il);
    aout[base + 16] = (_Float16)(o1[r] * il);
    aout[base + 32] = (_Float16)(o2[r] * il);
    aout[base + 48] = (_Float16)(o3[r] * il);
  }
}

// Final projection: out[n][o] = sum_c aout[n][c] * Wo[o][c], f32 output.
__global__ void __launch_bounds__(32) k_proj(const _Float16* __restrict__ ah,
                                             const _Float16* __restrict__ woh,
                                             float* __restrict__ out) {
  const int lane = threadIdx.x;
  const int n = lane & 15;
  const int mbase = (lane >> 4) * 8;
  const int tm = blockIdx.x;
  const int tn = blockIdx.y;
  const _Float16* arow = ah + (size_t)(tm * 16 + n) * kC;
  const _Float16* brow = woh + (size_t)(tn * 16 + n) * kC;
  v8f acc = {};
  for (int k = 0; k < kC; k += 32) {
    acc = wmma_f16(load_afrag(arow + k, lane), load_bfrag(brow + k, lane), acc);
  }
#pragma unroll
  for (int r = 0; r < 8; ++r) {
    out[(size_t)(tm * 16 + mbase + r) * kC + tn * 16 + n] = acc[r];
  }
}

extern "C" void kernel_launch(void* const* d_in, const int* in_sizes, int n_in,
                              void* d_out, int out_size, void* d_ws, size_t ws_size,
                              hipStream_t stream) {
  const float* x  = (const float*)d_in[0];
  const float* Wq = (const float*)d_in[1];
  const float* gq = (const float*)d_in[2];
  const float* Wk = (const float*)d_in[3];
  const float* gk = (const float*)d_in[4];
  const float* Wv = (const float*)d_in[5];
  const float* gv = (const float*)d_in[6];
  const float* Wo = (const float*)d_in[7];
  float* out = (float*)d_out;

  char* ws = (char*)d_ws;
  size_t off = 0;
  auto take = [&](size_t bytes) -> void* {
    void* p = ws + off;
    off += (bytes + 255) & ~(size_t)255;
    return p;
  };
  _Float16* xah  = (_Float16*)take((size_t)kN * kC * 2);
  float*    xsq  = (float*)take((size_t)kN * 4);
  _Float16* wqh  = (_Float16*)take((size_t)kC * kC * 2);
  _Float16* wkh  = (_Float16*)take((size_t)kC * kC * 2);
  _Float16* wvh  = (_Float16*)take((size_t)kC * kC * 2);
  _Float16* woh  = (_Float16*)take((size_t)kC * kC * 2);
  float*    wqsq = (float*)take((size_t)kC * 4);
  float*    wksq = (float*)take((size_t)kC * 4);
  float*    wvsq = (float*)take((size_t)kC * 4);
  float*    wosq = (float*)take((size_t)kC * 4);
  _Float16* qh   = (_Float16*)take((size_t)kB * kH * kT * kD * 2);
  _Float16* khp  = (_Float16*)take((size_t)kB * kH * kT * kD * 2);
  _Float16* vT   = (_Float16*)take((size_t)kB * kH * kT * kD * 2);
  _Float16* aout = (_Float16*)take((size_t)kN * kC * 2);

  k_prep<<<kN, 256, 0, stream>>>(x, xah, xsq, kC, 1);
  k_prep<<<kC, 256, 0, stream>>>(Wq, wqh, wqsq, kC, 1);
  k_prep<<<kC, 256, 0, stream>>>(Wk, wkh, wksq, kC, 1);
  k_prep<<<kC, 256, 0, stream>>>(Wv, wvh, wvsq, kC, 1);
  k_prep<<<kC, 256, 0, stream>>>(Wo, woh, wosq, kC, 0);

  dim3 gtile(kN / 16, kC / 16);
  k_qkv<<<gtile, 32, 0, stream>>>(xah, xsq, wqh, wqsq, gq, 0, qh);
  k_qkv<<<gtile, 32, 0, stream>>>(xah, xsq, wkh, wksq, gk, 1, khp);
  k_qkv<<<gtile, 32, 0, stream>>>(xah, xsq, wvh, wvsq, gv, 2, vT);

  k_attn<<<dim3(kT / 16, kH, kB), 32, 0, stream>>>(qh, khp, vT, aout);
  k_proj<<<gtile, 32, 0, stream>>>(aout, woh, out);

  (void)in_sizes; (void)n_in; (void)out_size; (void)ws_size;
}
